// GroupedQueryAttention_54546084659902
// MI455X (gfx1250) — matrix operand order repaired and checked by reading, not compiled, not run on hardware
//
#include <hip/hip_runtime.h>

// ---------------------------------------------------------------------------
// GQA for MI455X (gfx1250): f16 WMMA (v_wmma_f32_16x16x32_f16), fp32 accum.
// Stages:
//   1) gemm_tn<0> : Q proj  -> qh16  [B,H,S,64]   (f16)
//   2) gemm_tn<1> : K proj  -> kh16  [B,G,S,64]   (f16)
//   3) gemm_tn<2> : V proj  -> vT16  [B,G,64,S]   (f16, transposed for PV)
//   4) attn_kernel: fused QK^T -> softmax -> weights(fp32, NT store to d_out)
//                   -> PV -> ctx16 [B,S,2048] (f16)
//   5) gemm_tn<3> : dense   -> output fp32 (NT store) to d_out
// NT stores keep the 545 MB of write-once output traffic out of L2 so the
// re-read weight matrices / KV heads stay resident (192 MB global L2).
// ---------------------------------------------------------------------------

typedef __attribute__((ext_vector_type(16))) _Float16 v16h;
typedef __attribute__((ext_vector_type(8)))  _Float16 v8h;
typedef __attribute__((ext_vector_type(4)))  _Float16 v4h;
typedef __attribute__((ext_vector_type(8)))  float    v8f;

#define SHUF16(lo, hi) __builtin_shufflevector(lo, hi, 0,1,2,3,4,5,6,7,8,9,10,11,12,13,14,15)

// A-matrix fragment (16x32, M x K). Lane l<16: row l, K = {0..7, 16..23}+kbase.
// Lane l>=16: row l-16, K = {8..15, 24..31}+kbase.  Two 16B loads.
__device__ __forceinline__ v16h frag_a(const _Float16* row, int kbase, int lane_hi) {
  v8h lo = *(const v8h*)(row + kbase + lane_hi * 8);
  v8h hi = *(const v8h*)(row + kbase + 16 + lane_hi * 8);
  return SHUF16(lo, hi);
}

// B-matrix fragment (32x16, K x N). Lane n<16: col n, K = {0..7, 16..23}+kbase.
// Lane n>=16: col n-16, K = {8..15, 24..31}+kbase.  `row` points at the contiguous
// K-values of this lane's column (i.e., a row of W / K-proj / V^T).
__device__ __forceinline__ v16h frag_b(const _Float16* row, int kbase, int lane_hi) {
  v8h lo = *(const v8h*)(row + kbase + lane_hi * 8);
  v8h hi = *(const v8h*)(row + kbase + 16 + lane_hi * 8);
  return SHUF16(lo, hi);
}

__device__ __forceinline__ v8f wmma16(v16h a, v16h b, v8f c) {
  return __builtin_amdgcn_wmma_f32_16x16x32_f16(false, a, false, b, (short)0, c,
                                                false, false);
}

// ---------------------------------------------------------------------------
// Generic C = A @ W^T + bias,  A:[M,K] (fp32 or f16), W:[N,K] fp32.
// 64x64 tile per block, 8 waves, each wave owns two 16x16 output tiles.
// MODE 0: out f16, scatter to [B,32,S,64]      (Q heads)
// MODE 1: out f16, scatter to [B,8,S,64]       (K heads)
// MODE 2: out f16, scatter to [B,8,64,S]       (V heads, transposed)
// MODE 3: out fp32 row-major [M,2048]          (dense output, NT stores)
// ---------------------------------------------------------------------------
template <int MODE, bool AHALF>
__global__ __launch_bounds__(256) void gemm_tn(const void* __restrict__ Ap,
                                               const float* __restrict__ W,
                                               const float* __restrict__ bias,
                                               void* __restrict__ Cp, int K) {
  constexpr int LDSR = 72;  // 64 + 8-half pad (keeps every frag load 16B aligned)
  __shared__ _Float16 As[64 * LDSR];
  __shared__ _Float16 Bs[64 * LDSR];

  const int tid = threadIdx.x;
  const int wave = tid >> 5, lane = tid & 31;
  const int lane_lo = lane & 15, lane_hi = lane >> 4;
  const int n0 = blockIdx.x * 64;
  const int m0 = blockIdx.y * 64;
  const int mt = wave >> 1;           // wave's M-tile (0..3)
  const int nt0 = (wave & 1) * 2;     // wave's first N-tile (0 or 2)

  // staging coordinates (hoisted out of the k-loop)
  const int sr4 = tid >> 2, sc4 = (tid & 3) << 4;   // fp32 path: 4 rows/thread grp
  const int sr8 = tid >> 1, sc8 = (tid & 1) << 5;   // f16 path

  v8f acc0 = {}, acc1 = {};

  for (int k0 = 0; k0 < K; k0 += 64) {
    // ---- stage A tile (convert fp32 -> f16 on the fly, packed 8B stores) ----
    if (AHALF) {
      const _Float16* A = (const _Float16*)Ap;
      for (int i = tid; i < 64 * 8; i += 256) {
        int r = i >> 3, c = (i & 7) << 3;
        *(v8h*)(As + r * LDSR + c) = *(const v8h*)(A + (size_t)(m0 + r) * K + k0 + c);
      }
    } else {
      const float* A = (const float*)Ap;
      for (int i = tid; i < 64 * 16; i += 256) {
        int r = i >> 4, c = (i & 15) << 2;
        const float4 f = *(const float4*)(A + (size_t)(m0 + r) * K + k0 + c);
        v4h h = {(_Float16)f.x, (_Float16)f.y, (_Float16)f.z, (_Float16)f.w};
        *(v4h*)(As + r * LDSR + c) = h;
      }
    }
    // ---- stage W tile ----
    for (int i = tid; i < 64 * 16; i += 256) {
      int r = i >> 4, c = (i & 15) << 2;
      const float4 f = *(const float4*)(W + (size_t)(n0 + r) * K + k0 + c);
      v4h h = {(_Float16)f.x, (_Float16)f.y, (_Float16)f.z, (_Float16)f.w};
      *(v4h*)(Bs + r * LDSR + c) = h;
    }
    // prefetch next chunk while this chunk's WMMAs run (global_prefetch_b8)
    if (k0 + 64 < K) {
      __builtin_prefetch(W + (size_t)(n0 + sr4) * K + k0 + 64 + sc4, 0, 3);
      if (AHALF)
        __builtin_prefetch((const _Float16*)Ap + (size_t)(m0 + sr8) * K + k0 + 64 + sc8, 0, 3);
      else
        __builtin_prefetch((const float*)Ap + (size_t)(m0 + sr4) * K + k0 + 64 + sc4, 0, 3);
    }
    __syncthreads();

#pragma unroll
    for (int kk = 0; kk < 2; ++kk) {
      v16h a  = frag_a(As + (mt * 16 + lane_lo) * LDSR, kk * 32, lane_hi);
      v16h b0 = frag_b(Bs + ((nt0 + 0) * 16 + lane_lo) * LDSR, kk * 32, lane_hi);
      v16h b1 = frag_b(Bs + ((nt0 + 1) * 16 + lane_lo) * LDSR, kk * 32, lane_hi);
      acc0 = wmma16(a, b0, acc0);
      acc1 = wmma16(a, b1, acc1);
    }
    __syncthreads();
  }

  // ---- epilogue: bias + scatter ----
#pragma unroll
  for (int j = 0; j < 2; ++j) {
    v8f acc = j ? acc1 : acc0;
    const int nt = nt0 + j;
    const int col = n0 + nt * 16 + lane_lo;
    const float bv = bias[col];
#pragma unroll
    for (int r = 0; r < 8; ++r) {
      const int row = m0 + mt * 16 + r + 8 * lane_hi;
      const float v = acc[r] + bv;
      if (MODE == 3) {
        __builtin_nontemporal_store(v, (float*)Cp + (size_t)row * 2048 + col);
      } else {
        _Float16* C = (_Float16*)Cp;
        const int bI = row >> 10, s = row & 1023;
        const int hd = col >> 6, d = col & 63;
        size_t idx;
        if (MODE == 0)      idx = (((size_t)bI * 32 + hd) * 1024 + s) * 64 + d;
        else if (MODE == 1) idx = (((size_t)bI * 8 + hd) * 1024 + s) * 64 + d;
        else                idx = (((size_t)bI * 8 + hd) * 64 + d) * 1024 + s;
        C[idx] = (_Float16)v;
      }
    }
  }
}

// ---------------------------------------------------------------------------
// Fused attention. One block per (b, h, 16-query-row tile): 8192 blocks.
// Wave w owns logits columns [w*128, w*128+128) (8 WMMA accumulators, 64 VGPR).
// Cross-wave softmax via shuffles + LDS; normalized weights -> d_out (fp32,
// non-temporal); P (f16) -> LDS; PV with K partitioned across waves; 8-way
// LDS reduce -> ctx.
// ---------------------------------------------------------------------------
__global__ __launch_bounds__(256) void attn_kernel(
    const _Float16* __restrict__ qh, const _Float16* __restrict__ kh,
    const _Float16* __restrict__ vT, float* __restrict__ attnw,
    _Float16* __restrict__ ctx) {
  __shared__ _Float16 p16[16 * 1024];     // 32 KB: normalized probs, f16
  __shared__ _Float16 pctx[8 * 16 * 64];  // 16 KB: per-wave partial context
  __shared__ float sred[2][8][16];        // 1 KB : row max / row sum partials

  const int tid = threadIdx.x;
  const int wave = tid >> 5, lane = tid & 31;
  const int lane_lo = lane & 15, lane_hi = lane >> 4;

  const int qblk = blockIdx.x & 63;  // S/16 = 64 query tiles
  const int bh = blockIdx.x >> 6;    // 0..127 = b*32 + h
  const int b = bh >> 5, h = bh & 31;
  const int g = h >> 2;
  const int q0 = qblk * 16;

  const _Float16* Qb = qh + ((size_t)(b * 32 + h) * 1024 + q0) * 64;
  const _Float16* Kb = kh + (size_t)(b * 8 + g) * 1024 * 64;
  const _Float16* Vb = vT + (size_t)(b * 8 + g) * 64 * 1024;

  // ---- phase 1: logits = Q K^T * 1/8 for this wave's 128 columns ----
  const v16h aq0 = frag_a(Qb + lane_lo * 64, 0, lane_hi);
  const v16h aq1 = frag_a(Qb + lane_lo * 64, 32, lane_hi);

  v8f acc[8] = {};
#pragma unroll
  for (int t = 0; t < 8; ++t) {
    const _Float16* krow = Kb + (size_t)(wave * 128 + t * 16 + lane_lo) * 64;
    acc[t] = wmma16(aq0, frag_b(krow, 0, lane_hi), acc[t]);
    acc[t] = wmma16(aq1, frag_b(krow, 32, lane_hi), acc[t]);
  }
#pragma unroll
  for (int t = 0; t < 8; ++t)
#pragma unroll
    for (int r = 0; r < 8; ++r) acc[t][r] *= 0.125f;  // 1/sqrt(64)

  // ---- row max: per-lane over tiles, then lanes (16-groups), then waves ----
  float m[8];
#pragma unroll
  for (int r = 0; r < 8; ++r) {
    float v = acc[0][r];
#pragma unroll
    for (int t = 1; t < 8; ++t) v = fmaxf(v, acc[t][r]);
    v = fmaxf(v, __shfl_xor(v, 1, 32));
    v = fmaxf(v, __shfl_xor(v, 2, 32));
    v = fmaxf(v, __shfl_xor(v, 4, 32));
    v = fmaxf(v, __shfl_xor(v, 8, 32));
    m[r] = v;
  }
  if (lane_lo == 0)
    for (int r = 0; r < 8; ++r) sred[0][wave][lane_hi * 8 + r] = m[r];
  __syncthreads();
  float gm[8];
#pragma unroll
  for (int r = 0; r < 8; ++r) {
    float v = sred[0][0][lane_hi * 8 + r];
    for (int w2 = 1; w2 < 8; ++w2) v = fmaxf(v, sred[0][w2][lane_hi * 8 + r]);
    gm[r] = v;
  }

  // ---- exp + row sum ----
  float s[8];
#pragma unroll
  for (int r = 0; r < 8; ++r) s[r] = 0.f;
#pragma unroll
  for (int t = 0; t < 8; ++t)
#pragma unroll
    for (int r = 0; r < 8; ++r) {
      const float p = __expf(acc[t][r] - gm[r]);
      acc[t][r] = p;
      s[r] += p;
    }
#pragma unroll
  for (int r = 0; r < 8; ++r) {
    float v = s[r];
    v += __shfl_xor(v, 1, 32);
    v += __shfl_xor(v, 2, 32);
    v += __shfl_xor(v, 4, 32);
    v += __shfl_xor(v, 8, 32);
    s[r] = v;
  }
  if (lane_lo == 0)
    for (int r = 0; r < 8; ++r) sred[1][wave][lane_hi * 8 + r] = s[r];
  __syncthreads();
  float inv[8];
#pragma unroll
  for (int r = 0; r < 8; ++r) {
    float v = 0.f;
    for (int w2 = 0; w2 < 8; ++w2) v += sred[1][w2][lane_hi * 8 + r];
    inv[r] = 1.0f / v;
  }

  // ---- write normalized weights (fp32, non-temporal) and stage P (f16) ----
  float* Wout = attnw + ((size_t)bh * 1024 + q0) * 1024;
#pragma unroll
  for (int t = 0; t < 8; ++t) {
    const int colL = wave * 128 + t * 16 + lane_lo;
#pragma unroll
    for (int r = 0; r < 8; ++r) {
      const int rowL = r + 8 * lane_hi;
      const float p = acc[t][r] * inv[r];
      __builtin_nontemporal_store(p, Wout + (size_t)rowL * 1024 + colL);
      p16[rowL * 1024 + colL] = (_Float16)p;
    }
  }
  __syncthreads();

  // ---- phase 2: partial context over this wave's K-slice [wave*128, +128) ----
  v8f cacc[4] = {};
#pragma unroll
  for (int ks = 0; ks < 4; ++ks) {
    const int c0 = wave * 128 + ks * 32;
    const v16h a = frag_a(p16 + lane_lo * 1024, c0, lane_hi);
#pragma unroll
    for (int nt = 0; nt < 4; ++nt) {
      const v16h bv = frag_b(Vb + (size_t)(nt * 16 + lane_lo) * 1024, c0, lane_hi);
      cacc[nt] = wmma16(a, bv, cacc[nt]);
    }
  }
#pragma unroll
  for (int nt = 0; nt < 4; ++nt)
#pragma unroll
    for (int r = 0; r < 8; ++r)
      pctx[(wave * 16 + r + 8 * lane_hi) * 64 + nt * 16 + lane_lo] =
          (_Float16)cacc[nt][r];
  __syncthreads();

  // ---- 8-way reduce partials, write context [B,S,H*64] ----
  for (int i = tid; i < 16 * 64; i += 256) {
    const int row = i >> 6, col = i & 63;
    float v = 0.f;
#pragma unroll
    for (int w2 = 0; w2 < 8; ++w2) v += (float)pctx[(w2 * 16 + row) * 64 + col];
    ctx[((size_t)b * 1024 + q0 + row) * 2048 + h * 64 + col] = (_Float16)v;
  }
}

// ---------------------------------------------------------------------------
extern "C" void kernel_launch(void* const* d_in, const int* in_sizes, int n_in,
                              void* d_out, int out_size, void* d_ws,
                              size_t ws_size, hipStream_t stream) {
  const float* q       = (const float*)d_in[0];
  const float* k       = (const float*)d_in[1];
  const float* v       = (const float*)d_in[2];
  const float* wq_w    = (const float*)d_in[3];
  const float* wq_b    = (const float*)d_in[4];
  const float* wk_w    = (const float*)d_in[5];
  const float* wk_b    = (const float*)d_in[6];
  const float* wv_w    = (const float*)d_in[7];
  const float* wv_b    = (const float*)d_in[8];
  const float* dense_w = (const float*)d_in[9];
  const float* dense_b = (const float*)d_in[10];

  // workspace partition (halves): qh 8.39M | kh 2.10M | vT 2.10M | ctx 8.39M
  _Float16* qh  = (_Float16*)d_ws;
  _Float16* kh  = qh + (size_t)8388608;
  _Float16* vT  = kh + (size_t)2097152;
  _Float16* ctx = vT + (size_t)2097152;

  float* out   = (float*)d_out;
  float* attnw = out + (size_t)8388608;  // 4*1024*2048 output floats first

  dim3 blk(256);
  // Q/K/V projections (M=4096, K=2048; N=2048 / 512 / 512)
  gemm_tn<0, false><<<dim3(32, 64), blk, 0, stream>>>(q, wq_w, wq_b, qh, 2048);
  gemm_tn<1, false><<<dim3(8, 64),  blk, 0, stream>>>(k, wk_w, wk_b, kh, 2048);
  gemm_tn<2, false><<<dim3(8, 64),  blk, 0, stream>>>(v, wv_w, wv_b, vT, 2048);
  // Fused attention: 128 (b,h) pairs x 64 query tiles
  attn_kernel<<<8192, blk, 0, stream>>>(qh, kh, vT, attnw, ctx);
  // Output projection
  gemm_tn<3, true><<<dim3(32, 64), blk, 0, stream>>>(ctx, dense_w, dense_b, out,
                                                     2048);
}
